// WindowAttention_22823456211849
// MI455X (gfx1250) — matrix operand order repaired and checked by reading, not compiled, not run on hardware
//
#include <hip/hip_runtime.h>

#define HEADS 12
#define NTOK  49
#define DIMC  384

typedef __attribute__((ext_vector_type(16))) _Float16 v16h;
typedef __attribute__((ext_vector_type(8)))  float    v8f;

// ---- LDS layout (units: 16-bit elements) ----
#define XB_STRIDE 392                       // x / attn-out tile [64][392]
#define XB_OFF    0
#define XB_ELEMS  (64 * XB_STRIDE)          // 25088
#define QK_STRIDE 40                        // per-head Q/K [64][40]
#define QS_OFF    (XB_OFF + XB_ELEMS)       // 25088
#define QS_ELEMS  (HEADS * 64 * QK_STRIDE)  // 30720
#define KS_OFF    (QS_OFF + QS_ELEMS)       // 55808
#define VT_STRIDE 72                        // per-head V^T [32][72]
#define VT_OFF    (KS_OFF + QS_ELEMS)       // 86528
#define VT_ELEMS  (HEADS * 32 * VT_STRIDE)  // 27648
#define PB_OFF    (VT_OFF + VT_ELEMS)       // 114176  per-wave P [16][72]
#define PB_ELEMS  (8 * 16 * VT_STRIDE)      // 9216
#define SMEM_ELEMS (PB_OFF + PB_ELEMS)      // 123392 elems = 246784 B

__device__ __forceinline__ unsigned short f2h(float f) {
  _Float16 h = (_Float16)f;
  return __builtin_bit_cast(unsigned short, h);
}

// A-operand (16x32 f16): lane L -> row M=L%16, 16 contiguous K values starting at (L/16)*16
__device__ __forceinline__ v16h load_a16(const unsigned short* p) {
  union { v16h v; uint4 q[2]; } u;
  u.q[0] = *(const uint4*)(p);
  u.q[1] = *(const uint4*)(p + 8);
  return u.v;
}
// B-operand (32x16 f16, stored as B^T row-major): lane L -> row N=L%16,
// 16 contiguous K values starting at (L/16)*16
__device__ __forceinline__ v16h load_b16(const unsigned short* p) {
  union { v16h v; uint4 q[2]; } u;
  u.q[0] = *(const uint4*)(p);
  u.q[1] = *(const uint4*)(p + 8);
  return u.v;
}

#define WMMA_F16(A, B, C) \
  __builtin_amdgcn_wmma_f32_16x16x32_f16(false, (A), false, (B), (short)0, (C), false, false)

// ---------------- pre-kernels ----------------
// o[oo][ii] = w[ii][oo]  (w row stride = rows_out)
__global__ void prep_wT(const float* __restrict__ w, unsigned short* __restrict__ o,
                        int rows_out, int cols_out) {
  int i = blockIdx.x * 256 + threadIdx.x;
  if (i < rows_out * cols_out) {
    int oo = i / cols_out, ii = i - oo * cols_out;
    o[i] = f2h(w[(size_t)ii * rows_out + oo]);
  }
}

// padded bias [12][64][64]; columns >= 49 get -1e30 (softmax mask for free)
__global__ void prep_bias(const float* __restrict__ tbl, const int* __restrict__ ridx,
                          float* __restrict__ o) {
  int i = blockIdx.x * 256 + threadIdx.x;
  if (i < HEADS * 64 * 64) {
    int h = i >> 12, m = (i >> 6) & 63, n = i & 63;
    float v = 0.0f;
    if (n >= NTOK) v = -1e30f;
    else if (m < NTOK) v = tbl[(size_t)ridx[m * NTOK + n] * HEADS + h];
    o[i] = v;
  }
}

// ---------------- main kernel: one workgroup per window ----------------
__global__ __launch_bounds__(256) void win_attn_kernel(
    const float* __restrict__ x, const float* __restrict__ qkv_b,
    const float* __restrict__ proj_b,
    const unsigned short* __restrict__ wqkvT,   // [1152][384] f16
    const unsigned short* __restrict__ wprojT,  // [384][384]  f16
    const float* __restrict__ biasg,            // [12][64][64] f32
    float* __restrict__ out) {
  extern __shared__ unsigned short smem[];
  unsigned short* xb = smem + XB_OFF;   // x tile, later attn-out tile
  unsigned short* qs = smem + QS_OFF;
  unsigned short* ks = smem + KS_OFF;
  unsigned short* vt = smem + VT_OFF;
  unsigned short* pb = smem + PB_OFF;

  const int b    = blockIdx.x;
  const int tid  = threadIdx.x;
  const int lane = tid & 31;
  const int wave = tid >> 5;
  const int lr   = lane & 15;          // row/col-in-tile selector
  const int lg8  = (lane >> 4) * 8;    // accumulator (C-layout) row group
  const int lg16 = (lane >> 4) * 16;   // B-operand K group

  // ---- load x -> f16 LDS tile (rows >= 49 and col pad zeroed) ----
  for (int i = tid; i < XB_ELEMS / 4; i += 256)
    ((uint2*)xb)[i] = make_uint2(0u, 0u);
  __syncthreads();
  for (int idx = tid; idx < NTOK * (DIMC / 4); idx += 256) {
    int row = idx / 96, c4 = (idx - row * 96) * 4;
    const float4 f = *(const float4*)(x + ((size_t)b * NTOK + row) * DIMC + c4);
    uint2 o;
    o.x = (unsigned int)f2h(f.x) | ((unsigned int)f2h(f.y) << 16);
    o.y = (unsigned int)f2h(f.z) | ((unsigned int)f2h(f.w) << 16);
    *(uint2*)(xb + row * XB_STRIDE + c4) = o;
  }
  __syncthreads();

  // ---- Phase 1: QKV = x @ Wqkv + b ;  write Q,K row-major, V transposed ----
  for (int t = wave; t < 72; t += 8) {
    const int nc = t * 16;
    v8f acc[4] = {{}, {}, {}, {}};
    const unsigned short* wrow = wqkvT + (size_t)(nc + lr) * DIMC + lg16;
    for (int kb = 0; kb < DIMC; kb += 32) {
      __builtin_prefetch(wrow + kb + 32, 0, 1);
      v16h bt = load_b16(wrow + kb);
#pragma unroll
      for (int mi = 0; mi < 4; ++mi) {
        v16h at = load_a16(xb + (mi * 16 + lr) * XB_STRIDE + kb + lg16);
        acc[mi] = WMMA_F16(at, bt, acc[mi]);
      }
    }
    const float bias = qkv_b[nc + lr];
    const int sec  = nc / DIMC;
    const int rem  = nc - sec * DIMC;
    const int head = rem >> 5;
    const int fb   = rem & 31;         // 0 or 16 within the head
#pragma unroll
    for (int mi = 0; mi < 4; ++mi)
#pragma unroll
      for (int r = 0; r < 8; ++r) {
        unsigned short hv = f2h(acc[mi][r] + bias);
        int M = mi * 16 + r + lg8;     // token index (C-layout row)
        if (sec == 0)      qs[head * 64 * QK_STRIDE + M * QK_STRIDE + fb + lr] = hv;
        else if (sec == 1) ks[head * 64 * QK_STRIDE + M * QK_STRIDE + fb + lr] = hv;
        else               vt[head * 32 * VT_STRIDE + (fb + lr) * VT_STRIDE + M] = hv;
      }
  }
  __syncthreads();

  // ---- Phase 2: per-head attention (one wave per head) ----
  const float scale = 0.17677669529663687f;  // 1/sqrt(32)
  for (int h = wave; h < HEADS; h += 8) {
    v16h kbt[4];
#pragma unroll
    for (int nj = 0; nj < 4; ++nj)
      kbt[nj] = load_b16(ks + h * 64 * QK_STRIDE + (nj * 16 + lr) * QK_STRIDE + lg16);
    v16h vbt[2][2];
#pragma unroll
    for (int nf = 0; nf < 2; ++nf)
#pragma unroll
      for (int kk = 0; kk < 2; ++kk)
        vbt[nf][kk] = load_b16(vt + h * 32 * VT_STRIDE + (nf * 16 + lr) * VT_STRIDE +
                               kk * 32 + lg16);
    unsigned short* pw = pb + wave * 16 * VT_STRIDE;
    const float* bh = biasg + h * 64 * 64;

    for (int mi = 0; mi < 4; ++mi) {
      v16h qa = load_a16(qs + h * 64 * QK_STRIDE + (mi * 16 + lr) * QK_STRIDE + lg16);
      v8f s[4];
#pragma unroll
      for (int nj = 0; nj < 4; ++nj) {
        v8f z = {};
        s[nj] = WMMA_F16(qa, kbt[nj], z);
      }
      // scale + relative-position bias (padded cols carry -1e30 mask)
#pragma unroll
      for (int nj = 0; nj < 4; ++nj) {
        int n = nj * 16 + lr;
#pragma unroll
        for (int r = 0; r < 8; ++r) {
          int m = mi * 16 + r + lg8;
          s[nj][r] = s[nj][r] * scale + bh[m * 64 + n];
        }
      }
      // row softmax: rows live in 16-lane halves, reduce with xor 1,2,4,8
      v8f rm = s[0];
#pragma unroll
      for (int nj = 1; nj < 4; ++nj)
#pragma unroll
        for (int r = 0; r < 8; ++r) rm[r] = fmaxf(rm[r], s[nj][r]);
#pragma unroll
      for (int r = 0; r < 8; ++r) {
        float v = rm[r];
        v = fmaxf(v, __shfl_xor(v, 1, 32));
        v = fmaxf(v, __shfl_xor(v, 2, 32));
        v = fmaxf(v, __shfl_xor(v, 4, 32));
        v = fmaxf(v, __shfl_xor(v, 8, 32));
        rm[r] = v;
      }
      v8f sum = {};
#pragma unroll
      for (int nj = 0; nj < 4; ++nj)
#pragma unroll
        for (int r = 0; r < 8; ++r) {
          float e = __expf(s[nj][r] - rm[r]);
          s[nj][r] = e;
          sum[r] += e;
        }
#pragma unroll
      for (int r = 0; r < 8; ++r) {
        float v = sum[r];
        v += __shfl_xor(v, 1, 32);
        v += __shfl_xor(v, 2, 32);
        v += __shfl_xor(v, 4, 32);
        v += __shfl_xor(v, 8, 32);
        sum[r] = 1.0f / v;
      }
      // probs -> f16 via wave-private LDS (C-layout -> A-layout re-tile)
#pragma unroll
      for (int nj = 0; nj < 4; ++nj)
#pragma unroll
        for (int r = 0; r < 8; ++r)
          pw[(r + lg8) * VT_STRIDE + nj * 16 + lr] = f2h(s[nj][r] * sum[r]);
      // P @ V  (K = 64 tokens = 2 k-steps)
      v8f o[2] = {{}, {}};
#pragma unroll
      for (int kk = 0; kk < 2; ++kk) {
        v16h pa = load_a16(pw + lr * VT_STRIDE + kk * 32 + lg16);
#pragma unroll
        for (int nf = 0; nf < 2; ++nf)
          o[nf] = WMMA_F16(pa, vbt[nf][kk], o[nf]);
      }
      // attn-out f16 -> reuse xb tile (x no longer needed)
#pragma unroll
      for (int nf = 0; nf < 2; ++nf)
#pragma unroll
        for (int r = 0; r < 8; ++r)
          xb[(mi * 16 + r + lg8) * XB_STRIDE + h * 32 + nf * 16 + lr] = f2h(o[nf][r]);
    }
  }
  __syncthreads();

  // ---- Phase 3: out = attn_out @ Wproj + b ----
  for (int t = wave; t < 24; t += 8) {
    const int nc = t * 16;
    v8f acc[4] = {{}, {}, {}, {}};
    const unsigned short* wrow = wprojT + (size_t)(nc + lr) * DIMC + lg16;
    for (int kb = 0; kb < DIMC; kb += 32) {
      __builtin_prefetch(wrow + kb + 32, 0, 1);
      v16h bt = load_b16(wrow + kb);
#pragma unroll
      for (int mi = 0; mi < 4; ++mi) {
        v16h at = load_a16(xb + (mi * 16 + lr) * XB_STRIDE + kb + lg16);
        acc[mi] = WMMA_F16(at, bt, acc[mi]);
      }
    }
    const float pbias = proj_b[nc + lr];
#pragma unroll
    for (int mi = 0; mi < 4; ++mi)
#pragma unroll
      for (int r = 0; r < 8; ++r) {
        int m = mi * 16 + r + lg8;
        if (m < NTOK)
          out[((size_t)b * NTOK + m) * DIMC + nc + lr] = acc[mi][r] + pbias;
      }
  }
}

extern "C" void kernel_launch(void* const* d_in, const int* in_sizes, int n_in,
                              void* d_out, int out_size, void* d_ws, size_t ws_size,
                              hipStream_t stream) {
  const float* x      = (const float*)d_in[0];
  const float* qkv_w  = (const float*)d_in[1];
  const float* qkv_b  = (const float*)d_in[2];
  const float* proj_w = (const float*)d_in[3];
  const float* proj_b = (const float*)d_in[4];
  const float* rpb    = (const float*)d_in[5];
  const int*   ridx   = (const int*)d_in[6];

  unsigned short* wqkvT  = (unsigned short*)d_ws;                 // 1152*384 f16
  unsigned short* wprojT = wqkvT + (size_t)3 * DIMC * DIMC;       //  384*384 f16
  float*          biasg  = (float*)(wprojT + (size_t)DIMC * DIMC);// 12*64*64 f32

  prep_wT<<<(3 * DIMC * DIMC + 255) / 256, 256, 0, stream>>>(qkv_w, wqkvT, 3 * DIMC, DIMC);
  prep_wT<<<(DIMC * DIMC + 255) / 256, 256, 0, stream>>>(proj_w, wprojT, DIMC, DIMC);
  prep_bias<<<(HEADS * 64 * 64 + 255) / 256, 256, 0, stream>>>(rpb, ridx, biasg);

  const int B = in_sizes[0] / (NTOK * DIMC);
  const size_t smem_bytes = (size_t)SMEM_ELEMS * sizeof(unsigned short);
  win_attn_kernel<<<B, 256, smem_bytes, stream>>>(x, qkv_b, proj_b, wqkvT, wprojT,
                                                  biasg, (float*)d_out);
}
